// MultiHeadedAttention_9526237463137
// MI455X (gfx1250) — matrix operand order repaired and checked by reading, not compiled, not run on hardware
//
#include <hip/hip_runtime.h>

// ---------------------------------------------------------------------------
// MultiHeadedAttention forward for MI455X (gfx1250), wave32 + WMMA bf16.
// B=2, S=2048, D=1024, H=16, DK=64.
// ---------------------------------------------------------------------------

#define Bsz   2
#define Ssz   2048
#define Dsz   1024
#define Hsz   16
#define DKsz  64

typedef __attribute__((ext_vector_type(16))) __bf16 v16bf;
typedef __attribute__((ext_vector_type(8)))  float  v8f;
typedef int v4i_gv __attribute__((vector_size(16)));   // matches builtin param

union AFrag {                 // 16x32 bf16 A / 32x16 bf16 B operand (8 VGPRs)
    v16bf          v;
    unsigned short u[16];
    uint4          q[2];
};
union CFrag {                 // 16x16 f32 C/D (8 VGPRs)
    v8f   v;
    float f[8];
};

__device__ __forceinline__ unsigned short f2bf(float f) {
    union { float f; unsigned u; } c; c.f = f;
    unsigned u = c.u;
    unsigned r = (u + 0x7FFFu + ((u >> 16) & 1u)) >> 16;   // round-nearest-even
    return (unsigned short)r;
}

// ---------------------------------------------------------------------------
// Async global->LDS staging (CDNA5 GLOBAL_LOAD_ASYNC_TO_LDS_B128, ASYNCcnt).
// Diagnostic from probe round: param0 = AS1 (global) v4i*, param1 = AS3 (LDS)
// v4i*, then two integer immediates (offset, cpol).
// Guarded: falls back to load + ds_store if the builtins aren't declared.
// ---------------------------------------------------------------------------
#if defined(__has_builtin)
#if __has_builtin(__builtin_amdgcn_global_load_async_to_lds_b128) && \
    __has_builtin(__builtin_amdgcn_s_wait_asynccnt)
#define USE_ASYNC_LDS 1
#endif
#endif

__device__ __forceinline__ void copy16_g2l(const unsigned short* g,
                                           unsigned short* l) {
#ifdef USE_ASYNC_LDS
    __builtin_amdgcn_global_load_async_to_lds_b128(
        (__attribute__((address_space(1))) v4i_gv*)(g),
        (__attribute__((address_space(3))) v4i_gv*)(l), 0, 0);
#else
    *(uint4*)l = *(const uint4*)g;
#endif
}

__device__ __forceinline__ void wait_async_stage() {
#ifdef USE_ASYNC_LDS
    __builtin_amdgcn_s_wait_asynccnt(0);
#endif
}

// ---------------------------------------------------------------------------
// 1) f32 -> bf16 elementwise convert (2 elements/thread, packed dword store)
// ---------------------------------------------------------------------------
__global__ void cvt_bf16_kernel(const float* __restrict__ src,
                                unsigned short* __restrict__ dst, int n) {
    int i = (blockIdx.x * blockDim.x + threadIdx.x) * 2;
    if (i < n) {
        unsigned lo = f2bf(src[i]);
        unsigned hi = f2bf(src[i + 1]);
        ((unsigned*)dst)[i >> 1] = lo | (hi << 16);
    }
}

// ---------------------------------------------------------------------------
// 2) 1024x1024 f32 -> bf16 transpose (LDS 32x32 tile, bank-conflict padded)
// ---------------------------------------------------------------------------
__global__ void transpose_w_kernel(const float* __restrict__ W,
                                   unsigned short* __restrict__ Wt) {
    __shared__ float tile[32][33];
    int bx = blockIdx.x * 32, by = blockIdx.y * 32;
    int tx = threadIdx.x, ty = threadIdx.y;          // 32 x 8
#pragma unroll
    for (int r = 0; r < 32; r += 8)
        tile[ty + r][tx] = W[(size_t)(by + ty + r) * Dsz + bx + tx];
    __syncthreads();
#pragma unroll
    for (int r = 0; r < 32; r += 8)
        Wt[(size_t)(bx + ty + r) * Dsz + by + tx] = f2bf(tile[tx][ty + r]);
}

// ---------------------------------------------------------------------------
// 3) WMMA GEMM:  out[M=4096, N=1024] = X[bf16] * Wt^T[bf16] + bias
//    8 waves/WG, each wave: 16(M) x 64(N) tile, K loop of 32 (bf16 wmma K).
//    The 64x32 bf16 W^T tile (4 KB) is shared by all 8 waves -> staged in
//    LDS, double-buffered, filled with async global->LDS copies.
//    mode 0: Q -> [B,H,S,DK] bf16      mode 1: K -> [B,H,S,DK] bf16
//    mode 2: V -> Vt [B,H,DK,S] bf16   mode 3: f32 [B,S,D] (final output)
// ---------------------------------------------------------------------------
__global__ void gemm_bf16_kernel(const unsigned short* __restrict__ X,
                                 const unsigned short* __restrict__ Wt,
                                 const float* __restrict__ bias,
                                 void* __restrict__ outp, int mode) {
    __shared__ unsigned short Bt[2][64 * 32];        // double-buffered B tile

    const int tid  = threadIdx.x;
    const int lane = tid & 31;
    const int wave = tid >> 5;
    const int hi   = lane >> 4;       // half-wave select
    const int nc   = lane & 15;       // column within 16-wide tile
    const int m0   = blockIdx.x * 128 + wave * 16;
    const int n0   = blockIdx.y * 64;

    // cooperative staging: 256 threads x 16B = 64 rows x 64B (one 64x32 tile)
    const int srow  = tid >> 2;           // 0..63  (n within tile)
    const int skoff = (tid & 3) * 8;      // 0,8,16,24 (k within tile)
    const unsigned short* gstage = Wt + (size_t)(n0 + srow) * Dsz + skoff;

    CFrag acc[4] = {};
    const unsigned short* xrow = X + (size_t)(m0 + nc) * Dsz;   // A row = lane&15

    // prime buffer 0 with k-block 0
    copy16_g2l(gstage, &Bt[0][srow * 32 + skoff]);

    for (int kb = 0; kb < 32; ++kb) {
        const int cur = kb & 1;
        wait_async_stage();              // staged tile for this iter complete
        __syncthreads();                 // visible to all waves; prev reads done
        if (kb + 1 < 32)                 // prefetch next tile into other buffer
            copy16_g2l(gstage + (kb + 1) * 32,
                       &Bt[(kb + 1) & 1][srow * 32 + skoff]);

        AFrag a;
        a.q[0] = *(const uint4*)&xrow[kb * 32 + hi * 16];
        a.q[1] = *(const uint4*)&xrow[kb * 32 + hi * 16 + 8];
#pragma unroll
        for (int t = 0; t < 4; ++t) {
            AFrag b;
            const unsigned short* lb = &Bt[cur][(t * 16 + nc) * 32 + hi * 16];
            b.q[0] = *(const uint4*)lb;
            b.q[1] = *(const uint4*)(lb + 8);
            acc[t].v = __builtin_amdgcn_wmma_f32_16x16x32_bf16(
                false, a.v, false, b.v, (short)0, acc[t].v, false, false);
        }
    }

#pragma unroll
    for (int t = 0; t < 4; ++t) {
#pragma unroll
        for (int i = 0; i < 8; ++i) {
            int mg = m0 + i + 8 * hi;
            int ng = n0 + t * 16 + nc;
            float val = acc[t].f[i] + bias[ng];
            if (mode == 3) {
                ((float*)outp)[(size_t)mg * Dsz + ng] = val;
            } else {
                int b = mg >> 11, s = mg & (Ssz - 1);
                int h = ng >> 6,  dk = ng & (DKsz - 1);
                size_t idx;
                if (mode == 2)   // V transposed: [B,H,DK,S]
                    idx = (((size_t)b * Hsz + h) * DKsz + dk) * Ssz + s;
                else             // Q/K: [B,H,S,DK]
                    idx = (((size_t)b * Hsz + h) * Ssz + s) * DKsz + dk;
                ((unsigned short*)outp)[idx] = f2bf(val);
            }
        }
    }
}

// ---------------------------------------------------------------------------
// 4) Fused causal flash-attention, bf16 WMMA, online softmax.
//    grid: (S/128, B*H); 8 waves/WG; each wave: one 16-row Q tile.
//    (Per-wave trip counts diverge -> no workgroup-shared staging here.)
// ---------------------------------------------------------------------------
__global__ void attn_kernel(const unsigned short* __restrict__ Qh,
                            const unsigned short* __restrict__ Kh,
                            const unsigned short* __restrict__ Vt,
                            unsigned short* __restrict__ AO) {
    __shared__ unsigned short Pls[8][16 * 32];   // per-wave P transpose buffer

    const int lane  = threadIdx.x & 31;
    const int wave  = threadIdx.x >> 5;
    const int hi    = lane >> 4;
    const int nc    = lane & 15;
    const int bh    = blockIdx.y;                        // b*H + h
    const int qbase = blockIdx.x * 128 + wave * 16;

    const unsigned short* Qp = Qh + ((size_t)bh * Ssz) * DKsz;
    const unsigned short* Kp = Kh + ((size_t)bh * Ssz) * DKsz;
    const unsigned short* Vp = Vt + ((size_t)bh * DKsz) * Ssz;
    unsigned short* Pb = Pls[wave];

    // Q A-fragments: row = qbase + (lane&15), DK split 0..31 / 32..63
    AFrag aQ0, aQ1;
    {
        const unsigned short* qrow = Qp + (size_t)(qbase + nc) * DKsz;
        aQ0.q[0] = *(const uint4*)&qrow[hi * 16];
        aQ0.q[1] = *(const uint4*)&qrow[hi * 16 + 8];
        aQ1.q[0] = *(const uint4*)&qrow[32 + hi * 16];
        aQ1.q[1] = *(const uint4*)&qrow[32 + hi * 16 + 8];
    }

    CFrag o[4] = {};
    float mrow[8], lrow[8];
#pragma unroll
    for (int i = 0; i < 8; ++i) { mrow[i] = -1e30f; lrow[i] = 0.0f; }

    const float scale = 0.125f;                 // 1/sqrt(DK)
    const int nkb = (qbase + 47) >> 5;          // causal: 32-key blocks needed

    for (int kb = 0; kb < nkb; ++kb) {
        const int kstart = kb * 32;
        const bool full  = (kstart + 31) <= qbase;

        // prefetch next K/V blocks into the cache hierarchy
        if (kb + 1 < nkb) {
            __builtin_prefetch(Kp + (size_t)(kstart + 32 + nc) * DKsz, 0, 1);
            __builtin_prefetch(Vp + (size_t)nc * Ssz + kstart + 32, 0, 1);
        }

        // ---- scores S = Q K^T : two 16x16 C tiles (keys +0..15, +16..31)
        CFrag c0 = {}, c1 = {};
        {
            const unsigned short* k0 = Kp + (size_t)(kstart + nc) * DKsz + hi * 16;
            const unsigned short* k1 = k0 + 16 * DKsz;
            AFrag b00, b01, b10, b11;
            b00.q[0] = *(const uint4*)k0;        b00.q[1] = *(const uint4*)(k0 + 8);
            b01.q[0] = *(const uint4*)(k0 + 32); b01.q[1] = *(const uint4*)(k0 + 40);
            b10.q[0] = *(const uint4*)k1;        b10.q[1] = *(const uint4*)(k1 + 8);
            b11.q[0] = *(const uint4*)(k1 + 32); b11.q[1] = *(const uint4*)(k1 + 40);
            c0.v = __builtin_amdgcn_wmma_f32_16x16x32_bf16(false, aQ0.v, false, b00.v, (short)0, c0.v, false, false);
            c0.v = __builtin_amdgcn_wmma_f32_16x16x32_bf16(false, aQ1.v, false, b01.v, (short)0, c0.v, false, false);
            c1.v = __builtin_amdgcn_wmma_f32_16x16x32_bf16(false, aQ0.v, false, b10.v, (short)0, c1.v, false, false);
            c1.v = __builtin_amdgcn_wmma_f32_16x16x32_bf16(false, aQ1.v, false, b11.v, (short)0, c1.v, false, false);
        }

        // ---- online softmax update (row-wise over 32 keys)
#pragma unroll
        for (int i = 0; i < 8; ++i) {
            float s0 = c0.f[i] * scale;
            float s1 = c1.f[i] * scale;
            if (!full) {
                int qrow = qbase + i + 8 * hi;
                if (kstart + nc      > qrow) s0 = -1e9f;
                if (kstart + 16 + nc > qrow) s1 = -1e9f;
            }
            float r = fmaxf(s0, s1);
#pragma unroll
            for (int off = 8; off >= 1; off >>= 1)
                r = fmaxf(r, __shfl_xor(r, off, 32));
            float mnew  = fmaxf(mrow[i], r);
            float alpha = __expf(mrow[i] - mnew);
            float p0    = __expf(s0 - mnew);
            float p1    = __expf(s1 - mnew);
            float ps    = p0 + p1;
#pragma unroll
            for (int off = 8; off >= 1; off >>= 1)
                ps += __shfl_xor(ps, off, 32);
            lrow[i] = lrow[i] * alpha + ps;
            mrow[i] = mnew;
#pragma unroll
            for (int t = 0; t < 4; ++t) o[t].f[i] *= alpha;

            // stage P (bf16) for A-fragment transpose
            int m = i + 8 * hi;
            Pb[m * 32 + nc]      = f2bf(p0);
            Pb[m * 32 + 16 + nc] = f2bf(p1);
        }

        // ---- P as 16x32 A-fragment (wave-local LDS, DS ops in-order per wave)
        AFrag aP;
        aP.q[0] = *(const uint4*)&Pb[nc * 32 + hi * 16];
        aP.q[1] = *(const uint4*)&Pb[nc * 32 + hi * 16 + 8];

        // ---- O += P V : Vt rows are contiguous in key dim
#pragma unroll
        for (int t = 0; t < 4; ++t) {
            AFrag bV;
            const unsigned short* vrow =
                Vp + (size_t)(t * 16 + nc) * Ssz + kstart + hi * 16;
            bV.q[0] = *(const uint4*)vrow;
            bV.q[1] = *(const uint4*)(vrow + 8);
            o[t].v = __builtin_amdgcn_wmma_f32_16x16x32_bf16(
                false, aP.v, false, bV.v, (short)0, o[t].v, false, false);
        }
    }

    // ---- epilogue: normalize, merge heads -> [B,S,D] bf16
    const int b = bh >> 4, h = bh & 15;
#pragma unroll
    for (int i = 0; i < 8; ++i) {
        float inv = 1.0f / lrow[i];
        int srow  = qbase + i + 8 * hi;
        size_t base = ((size_t)b * Ssz + srow) * Dsz + h * DKsz;
#pragma unroll
        for (int t = 0; t < 4; ++t)
            AO[base + t * 16 + nc] = f2bf(o[t].f[i] * inv);
    }
}

// ---------------------------------------------------------------------------
// Host-side launch. Workspace layout (bf16 unless noted), ~64 MB total:
//   Xq/Xk/Xv (8 MB ea) | Wq/Wk/Wv/Wo^T (2 MB ea) | Qh/Kh/Vt (8 MB ea) | AO (8 MB)
// ---------------------------------------------------------------------------
extern "C" void kernel_launch(void* const* d_in, const int* in_sizes, int n_in,
                              void* d_out, int out_size, void* d_ws, size_t ws_size,
                              hipStream_t stream) {
    const float* q  = (const float*)d_in[0];
    const float* k  = (const float*)d_in[1];
    const float* v  = (const float*)d_in[2];
    //               d_in[3] = mask (tril) -- reproduced from indices on device
    const float* Wq = (const float*)d_in[4];
    const float* bq = (const float*)d_in[5];
    const float* Wk = (const float*)d_in[6];
    const float* bk = (const float*)d_in[7];
    const float* Wv = (const float*)d_in[8];
    const float* bv = (const float*)d_in[9];
    const float* Wo = (const float*)d_in[10];
    const float* bo = (const float*)d_in[11];

    char* ws = (char*)d_ws;
    const size_t ACT = (size_t)Bsz * Ssz * Dsz * 2;   // 8 MB (bf16 activations)
    const size_t WMT = (size_t)Dsz * Dsz * 2;         // 2 MB (bf16 weights)
    unsigned short* Xq  = (unsigned short*)(ws);
    unsigned short* Xk  = (unsigned short*)(ws + ACT);
    unsigned short* Xv  = (unsigned short*)(ws + 2 * ACT);
    unsigned short* Wqt = (unsigned short*)(ws + 3 * ACT);
    unsigned short* Wkt = (unsigned short*)(ws + 3 * ACT + WMT);
    unsigned short* Wvt = (unsigned short*)(ws + 3 * ACT + 2 * WMT);
    unsigned short* Wot = (unsigned short*)(ws + 3 * ACT + 3 * WMT);
    unsigned short* Qh  = (unsigned short*)(ws + 3 * ACT + 4 * WMT);
    unsigned short* Kh  = (unsigned short*)(ws + 4 * ACT + 4 * WMT);
    unsigned short* Vt  = (unsigned short*)(ws + 5 * ACT + 4 * WMT);
    unsigned short* AO  = (unsigned short*)(ws + 6 * ACT + 4 * WMT);

    const int nAct = Bsz * Ssz * Dsz;                 // 4M elements

    // 1) activations -> bf16
    cvt_bf16_kernel<<<nAct / 512, 256, 0, stream>>>(q, Xq, nAct);
    cvt_bf16_kernel<<<nAct / 512, 256, 0, stream>>>(k, Xk, nAct);
    cvt_bf16_kernel<<<nAct / 512, 256, 0, stream>>>(v, Xv, nAct);

    // 2) weights -> bf16 transposed
    dim3 tg(32, 32), tb(32, 8);
    transpose_w_kernel<<<tg, tb, 0, stream>>>(Wq, Wqt);
    transpose_w_kernel<<<tg, tb, 0, stream>>>(Wk, Wkt);
    transpose_w_kernel<<<tg, tb, 0, stream>>>(Wv, Wvt);
    transpose_w_kernel<<<tg, tb, 0, stream>>>(Wo, Wot);

    // 3) QKV projections (WMMA), head-split layouts
    dim3 gg(32, 16);   // M/128 x N/64
    gemm_bf16_kernel<<<gg, 256, 0, stream>>>(Xq, Wqt, bq, Qh, 0);
    gemm_bf16_kernel<<<gg, 256, 0, stream>>>(Xk, Wkt, bk, Kh, 1);
    gemm_bf16_kernel<<<gg, 256, 0, stream>>>(Xv, Wvt, bv, Vt, 2);

    // 4) fused causal attention
    attn_kernel<<<dim3(Ssz / 128, Bsz * Hsz), 256, 0, stream>>>(Qh, Kh, Vt, AO);

    // 5) output projection -> f32 d_out
    gemm_bf16_kernel<<<gg, 256, 0, stream>>>(AO, Wot, bo, d_out, 3);
}
